// Network_51599737094896
// MI455X (gfx1250) — matrix operand order repaired and checked by reading, not compiled, not run on hardware
//
#include <hip/hip_runtime.h>
#include <stddef.h>
#include <stdint.h>

// ---------------------------------------------------------------------------
// Types for CDNA5 WMMA (wave32): D = A(16x32 bf16) * B(32x16 bf16) + C(16x16 f32)
// ---------------------------------------------------------------------------
typedef __attribute__((ext_vector_type(16))) __bf16 v16bf;
typedef __attribute__((ext_vector_type(8)))  __bf16 v8bf;
typedef __attribute__((ext_vector_type(8)))  float  v8f;
typedef __attribute__((ext_vector_type(4)))  unsigned v4u;
typedef __attribute__((ext_vector_type(8)))  int      v8i;
typedef __attribute__((ext_vector_type(4)))  int      v4i;

#if __has_builtin(__builtin_amdgcn_tensor_load_to_lds) && \
    __has_builtin(__builtin_amdgcn_s_wait_tensorcnt)
#define HAVE_TDM 1
#else
#define HAVE_TDM 0
#endif

union FragBF { v16bf f; v8bf h[2]; };

static __device__ inline v8bf ldv8(const __bf16* p) {
    return *reinterpret_cast<const v8bf*>(p);
}
static __device__ inline __bf16 f2bf(float x) {
    unsigned u = __builtin_bit_cast(unsigned, x);
    u += 0x7fffu + ((u >> 16) & 1u);              // round-to-nearest-even
    unsigned short s = (unsigned short)(u >> 16);
    return __builtin_bit_cast(__bf16, s);
}
static __device__ inline float bf2f(__bf16 b) {
    unsigned short s = __builtin_bit_cast(unsigned short, b);
    unsigned u = ((unsigned)s) << 16;
    return __builtin_bit_cast(float, u);
}
static __device__ inline v8f wmma_bf16(const FragBF& a, const FragBF& b, v8f c) {
    return __builtin_amdgcn_wmma_f32_16x16x32_bf16(false, a.f, false, b.f,
                                                   (short)0, c, false, false);
}
#define V8F_ZERO v8f{0.f,0.f,0.f,0.f,0.f,0.f,0.f,0.f}

#define NN     4096
#define PIX    81
#define E_DIM  256

// ---------------------------------------------------------------------------
// Weight pre-pack: f32 [rows][K] row-major -> bf16 [rows][Kpad], zero padded
// ---------------------------------------------------------------------------
__global__ __launch_bounds__(256) void pack_w_k(const float* __restrict__ w,
                                                __bf16* __restrict__ o,
                                                int rows, int K, int Kpad) {
    int idx = blockIdx.x * 256 + threadIdx.x;
    int total = rows * Kpad;
    if (idx >= total) return;
    int r = idx / Kpad, k = idx - r * Kpad;
    float v = (k < K) ? w[(size_t)r * K + k] : 0.f;
    o[idx] = f2bf(v);
}

__global__ __launch_bounds__(256) void cvt_bf_k(const float* __restrict__ s,
                                                __bf16* __restrict__ d, int n) {
    int idx = blockIdx.x * 256 + threadIdx.x;
    if (idx < n) d[idx] = f2bf(s[idx]);
}

// ---------------------------------------------------------------------------
// Implicit-GEMM 3x3 conv, pad=1, on 9x9 maps. One image per block, 256 thr.
// out[n][co][p] (co=128) = relu?(bias + skip? + sum_k W[co][k]*im2col[k][p])
// A = packed weights (global), B = im2col tile built in LDS per K-step.
// ---------------------------------------------------------------------------
template<int CIN, int KPAD, bool IN_F32, bool HAS_SKIP, bool RELU>
__global__ __launch_bounds__(256) void conv3x3_k(const void* __restrict__ inp,
                                                 const __bf16* __restrict__ wpk,
                                                 const float* __restrict__ bias,
                                                 const __bf16* __restrict__ skip,
                                                 __bf16* __restrict__ out) {
    const int n = blockIdx.x;
    const int tid = threadIdx.x;
    const int wave = tid >> 5, lane = tid & 31;
    const bool lh = lane < 16;
    const int lm = lane & 15;
    const int K = CIN * 9;

    __shared__ __align__(16) __bf16 in_lds[CIN * 121];   // 11x11 zero-halo per ch
    __shared__ __align__(16) __bf16 B_lds[96 * 32];      // [pixel][kk]

    for (int i = tid; i < CIN * 121; i += 256) in_lds[i] = f2bf(0.f);
    __syncthreads();
    for (int i = tid; i < CIN * PIX; i += 256) {
        int ci = i / PIX, p = i - ci * PIX;
        float v;
        if (IN_F32) v = ((const float*)inp)[(size_t)n * CIN * PIX + i];
        else        v = bf2f(((const __bf16*)inp)[(size_t)n * CIN * PIX + i]);
        int hh = p / 9, ww = p - hh * 9;
        in_lds[ci * 121 + (hh + 1) * 11 + (ww + 1)] = f2bf(v);
    }

    v8f acc[6];
#pragma unroll
    for (int pt = 0; pt < 6; ++pt) acc[pt] = V8F_ZERO;
    const __bf16 zbf = f2bf(0.f);

    for (int kb = 0; kb < KPAD; kb += 32) {
        __syncthreads();
        // build B_lds[p][kk] = im2col[kb+kk][p]
#pragma unroll
        for (int e = 0; e < 12; ++e) {
            int lin = tid + 256 * e;
            int kk = lin & 31, p = lin >> 5;
            int k = kb + kk;
            __bf16 bv = zbf;
            if (k < K && p < PIX) {
                int ci = k / 9, rr = k - ci * 9;
                int kh = rr / 3, kw = rr - kh * 3;
                int hh = p / 9, ww = p - hh * 9;
                bv = in_lds[ci * 121 + (hh + kh) * 11 + (ww + kw)];
            }
            B_lds[p * 32 + kk] = bv;
        }
        __syncthreads();

        FragBF a;
        {
            int co = wave * 16 + lm;
            size_t base = (size_t)co * KPAD + kb + (lh ? 0 : 8);
            a.h[0] = ldv8(wpk + base);
            a.h[1] = ldv8(wpk + base + 16);
            if (kb + 32 < KPAD)                       // -> global_prefetch_b8
                __builtin_prefetch((const void*)(wpk + base + 32), 0, 0);
        }
        const int boff = lh ? 0 : 8;
#pragma unroll
        for (int pt = 0; pt < 6; ++pt) {
            FragBF b;
            const __bf16* bp = &B_lds[(pt * 16 + lm) * 32 + boff];
            b.h[0] = ldv8(bp);
            b.h[1] = ldv8(bp + 16);
            acc[pt] = wmma_bf16(a, b, acc[pt]);
        }
    }

#pragma unroll
    for (int pt = 0; pt < 6; ++pt) {
        int p = pt * 16 + lm;
        if (p >= PIX) continue;
#pragma unroll
        for (int r = 0; r < 8; ++r) {
            int co = wave * 16 + r + (lh ? 0 : 8);
            float v = acc[pt][r] + bias[co];
            size_t oidx = ((size_t)n * 128 + co) * PIX + p;
            if (HAS_SKIP) v += bf2f(skip[oidx]);
            if (RELU) v = v > 0.f ? v : 0.f;
            out[oidx] = f2bf(v);
        }
    }
}

// ---------------------------------------------------------------------------
// Generic fully-connected WMMA GEMM: out[row][o] = eps(bias + skip? +
//   sum_k A[row][k] * W[o][k]).  Block: 64 rows x 128 outs, 8 waves.
// ---------------------------------------------------------------------------
__global__ __launch_bounds__(256) void linear_k(const float* __restrict__ A, int lda,
                                                int Kc, int Kpad,
                                                const __bf16* __restrict__ wpk,
                                                const float* __restrict__ bias, int Nout,
                                                const float* __restrict__ skip, int ldskip,
                                                float* __restrict__ out, int ldo,
                                                int relu) {
    const int mb = blockIdx.x * 64;
    const int tid = threadIdx.x;
    const int wave = tid >> 5, lane = tid & 31;
    const bool lh = lane < 16;
    const int lm = lane & 15;
    const int ob = blockIdx.y * 128 + wave * 16;
    const bool active = ob < Nout;

    __shared__ __align__(16) __bf16 A_lds[64 * 32];

    v8f acc[4];
#pragma unroll
    for (int i = 0; i < 4; ++i) acc[i] = V8F_ZERO;

    for (int kb = 0; kb < Kpad; kb += 32) {
        __syncthreads();
#pragma unroll
        for (int e = 0; e < 8; ++e) {
            int lin = tid + 256 * e;
            int kk = lin & 31, m = lin >> 5;
            int k = kb + kk;
            float v = (k < Kc) ? A[(size_t)(mb + m) * lda + k] : 0.f;
            A_lds[m * 32 + kk] = f2bf(v);
        }
        __syncthreads();
        if (active) {
            FragBF b;
            size_t wb = (size_t)(ob + lm) * Kpad + kb + (lh ? 0 : 16);
            b.h[0] = ldv8(wpk + wb);
            b.h[1] = ldv8(wpk + wb + 8);
            if (kb + 32 < Kpad)                       // -> global_prefetch_b8
                __builtin_prefetch((const void*)(wpk + wb + 32), 0, 0);
            const int aoff = lh ? 0 : 16;
#pragma unroll
            for (int ms = 0; ms < 4; ++ms) {
                FragBF a;
                const __bf16* ap = &A_lds[(ms * 16 + lm) * 32 + aoff];
                a.h[0] = ldv8(ap);
                a.h[1] = ldv8(ap + 8);
                acc[ms] = wmma_bf16(a, b, acc[ms]);
            }
        }
    }
    if (!active) return;
#pragma unroll
    for (int ms = 0; ms < 4; ++ms) {
#pragma unroll
        for (int r = 0; r < 8; ++r) {
            int row = mb + ms * 16 + r + (lh ? 0 : 8);
            int o = ob + lm;
            float v = acc[ms][r] + bias[o];
            if (skip) v += skip[(size_t)row * ldskip + o];
            if (relu) v = v > 0.f ? v : 0.f;
            out[(size_t)row * ldo + o] = v;
        }
    }
}

// ---------------------------------------------------------------------------
// conv2 (1x1, 128->8) + relu.  out x2[n][c*81+p] f32.
// ---------------------------------------------------------------------------
__global__ __launch_bounds__(256) void conv2_k(const __bf16* __restrict__ x,
                                               const float* __restrict__ w2,
                                               const float* __restrict__ b2,
                                               float* __restrict__ x2) {
    const int n = blockIdx.x;
    const int tid = threadIdx.x;
    __shared__ __bf16 xl[128 * PIX];
    __shared__ float  wl[8 * 128];
    for (int i = tid; i < 128 * PIX; i += 256)
        xl[i] = x[(size_t)n * 128 * PIX + i];
    for (int i = tid; i < 8 * 128; i += 256) wl[i] = w2[i];
    __syncthreads();
    for (int o = tid; o < 8 * PIX; o += 256) {
        int c = o / PIX, p = o - c * PIX;
        float s = b2[c];
        for (int ci = 0; ci < 128; ++ci)
            s += wl[c * 128 + ci] * bf2f(xl[ci * PIX + p]);
        x2[(size_t)n * (8 * PIX) + o] = s > 0.f ? s : 0.f;
    }
}

// ---------------------------------------------------------------------------
// GRU gate math with h0 == 0 (gh reduces to bhh): h = (1-z)*n
// ---------------------------------------------------------------------------
__global__ __launch_bounds__(256) void gru_k(const float* __restrict__ gx,
                                             const float* __restrict__ bhh,
                                             float* __restrict__ h, int total) {
    int idx = blockIdx.x * 256 + threadIdx.x;
    if (idx >= total) return;
    int n = idx >> 8, e = idx & 255;
    const float* g = gx + (size_t)n * 768;
    float r  = 1.f / (1.f + __expf(-(g[e]       + bhh[e])));
    float z  = 1.f / (1.f + __expf(-(g[256 + e] + bhh[256 + e])));
    float nn = tanhf(g[512 + e] + r * bhh[512 + e]);
    h[idx] = (1.f - z) * nn;
}

// ---------------------------------------------------------------------------
// Mask precompute.  mask[i,m] = adj[i,m] && m in {rank_i(0),rank_i(1),rank_i(2)}
// (stable argsort semantics).  Stores 3 masked column indices (or -1) per row.
// ---------------------------------------------------------------------------
__global__ __launch_bounds__(256) void mask_k(const float* __restrict__ pos,
                                              int* __restrict__ mc) {
    const int i = blockIdx.x;
    const int tid = threadIdx.x;
    const float ax = pos[i * 4], ay = pos[i * 4 + 1];
    float dj[3];
#pragma unroll
    for (int j = 0; j < 3; ++j) {
        float dx = ax - pos[j * 4], dy = ay - pos[j * 4 + 1];
        dj[j] = sqrtf(dx * dx + dy * dy);
    }
    __shared__ int scnt[3];
    if (tid < 3) scnt[tid] = 0;
    __syncthreads();
    int lc[3] = {0, 0, 0};
    for (int m = tid; m < NN; m += 256) {
        float dx = ax - pos[m * 4], dy = ay - pos[m * 4 + 1];
        float d = sqrtf(dx * dx + dy * dy);
#pragma unroll
        for (int j = 0; j < 3; ++j)
            if (d < dj[j] || (d == dj[j] && m < j)) lc[j]++;
    }
#pragma unroll
    for (int j = 0; j < 3; ++j) atomicAdd(&scnt[j], lc[j]);
    __syncthreads();
    if (tid == 0) {
#pragma unroll
        for (int j = 0; j < 3; ++j) {
            int r = scnt[j];
            float dx = ax - pos[r * 4], dy = ay - pos[r * 4 + 1];
            bool adj = (dx <= 4.0f) && (dy <= 4.0f);
            mc[i * 4 + j] = adj ? r : -1;
        }
        mc[i * 4 + 3] = -1;
    }
}

// ---------------------------------------------------------------------------
// Flash attention, 4 heads, DH=64, mask = -1e9 at <=3 columns per row.
// Block = (64 q rows, head). Wave w: qsub = w&3 (16 rows), dhalf = w>>2 (32 d).
// K tile staged via Tensor Data Mover when available (2D tile 32x64 bf16,
// row stride 768 elements), else vector loads.
// ---------------------------------------------------------------------------
__global__ __launch_bounds__(256) void attn_k(const __bf16* __restrict__ qkv,
                                              const int* __restrict__ mc,
                                              float* __restrict__ ao) {
    const int qb = blockIdx.x * 64;
    const int hh = blockIdx.y;
    const int tid = threadIdx.x;
    const int wave = tid >> 5, lane = tid & 31;
    const bool lh = lane < 16;
    const int lm = lane & 15;
    const int qsub = wave & 3, dhalf = wave >> 2;

    __shared__ __align__(16) __bf16 Kl[32 * 64];   // [m][d]
    __shared__ __align__(16) __bf16 Vl[64 * 32];   // [d][m]
    __shared__ __align__(16) float  Sl[64 * 32];
    __shared__ __align__(16) __bf16 Pl[64 * 32];
    __shared__ float rowM[64], rowSum[64], rowScale[64];
    __shared__ int   mcl[64][3];

    FragBF aQ[2];
    {
        int row = qb + qsub * 16 + lm;
        const __bf16* qp = qkv + (size_t)row * 768 + hh * 64;
        int o0 = lh ? 0 : 8;
        aQ[0].h[0] = ldv8(qp + o0);       aQ[0].h[1] = ldv8(qp + o0 + 16);
        aQ[1].h[0] = ldv8(qp + 32 + o0);  aQ[1].h[1] = ldv8(qp + 32 + o0 + 16);
    }
    v8f oa0 = V8F_ZERO, oa1 = V8F_ZERO;
    if (tid < 64) {
        rowM[tid] = -3.0e38f;
        rowSum[tid] = 0.f;
        int grow = qb + tid;
        mcl[tid][0] = mc[grow * 4 + 0];
        mcl[tid][1] = mc[grow * 4 + 1];
        mcl[tid][2] = mc[grow * 4 + 2];
    }

    for (int mb2 = 0; mb2 < NN; mb2 += 32) {
        __syncthreads();
#if HAVE_TDM
        if (wave == 0) {   // TDM: 2D tile, 32 rows x 64 bf16, stride 768 elems
            unsigned long long ga =
                (unsigned long long)(uintptr_t)(qkv + (size_t)mb2 * 768 + 256 + hh * 64);
            unsigned lds_off = (unsigned)(uintptr_t)(void*)&Kl[0];
            v4u g0 = { 1u,                               // count=1 (valid), load
                       lds_off,                          // D#.lds_addr
                       (unsigned)ga,                     // global_addr[31:0]
                       (unsigned)((ga >> 32) & 0x01FFFFFFu) | (2u << 30) }; // type=2
            v8i g1 = { (int)(1u << 16),                  // data_size=2B, wg_mask=0
                       (int)(64u << 16),                 // tensor_dim0=64
                       (int)(32u << 16),                 // tensor_dim1=32
                       (int)(64u << 16),                 // tile_dim0=64
                       32,                               // tile_dim1=32
                       768,                              // tensor_dim0_stride=768
                       0, 0 };
            v4i gz = { 0, 0, 0, 0 };
            v8i gz8 = { 0, 0, 0, 0, 0, 0, 0, 0 };
            __builtin_amdgcn_tensor_load_to_lds(g0, g1, gz, gz, gz8, 0);
            __builtin_amdgcn_s_wait_tensorcnt(0);
        }
#else
        {   // stage K chunk [32][64]
            int m_ = tid >> 3, dp = (tid & 7) * 8;
            *reinterpret_cast<v8bf*>(&Kl[m_ * 64 + dp]) =
                ldv8(qkv + (size_t)(mb2 + m_) * 768 + 256 + hh * 64 + dp);
        }
#endif
        {   // stage V chunk transposed -> [d][m]
            int base = tid * 8;
            int m_ = base >> 6, d0 = base & 63;
            v8bf vv = ldv8(qkv + (size_t)(mb2 + m_) * 768 + 512 + hh * 64 + d0);
#pragma unroll
            for (int e2 = 0; e2 < 8; ++e2) Vl[(d0 + e2) * 32 + m_] = vv[e2];
        }
        __syncthreads();

        if (dhalf == 0) {   // S = Q K^T for this wave's 16 q rows x 32 cols
#pragma unroll
            for (int msub = 0; msub < 2; ++msub) {
                v8f s = V8F_ZERO;
#pragma unroll
                for (int kb2 = 0; kb2 < 2; ++kb2) {
                    FragBF bK;
                    const __bf16* kp = &Kl[(msub * 16 + lm) * 64 + kb2 * 32 + (lh ? 0 : 8)];
                    bK.h[0] = ldv8(kp);
                    bK.h[1] = ldv8(kp + 16);
                    s = wmma_bf16(aQ[kb2], bK, s);
                }
#pragma unroll
                for (int r = 0; r < 8; ++r) {
                    int lr = qsub * 16 + r + (lh ? 0 : 8);
                    Sl[lr * 32 + msub * 16 + lm] = s[r] * 0.125f;
                }
            }
        }
        __syncthreads();

        if (tid < 64) {     // per-row online softmax over this 32-col chunk
            int row = tid;
            int r0 = mcl[row][0], r1 = mcl[row][1], r2 = mcl[row][2];
            float mOld = rowM[row], cmax = -3.0e38f;
            float sv[32];
#pragma unroll
            for (int c = 0; c < 32; ++c) {
                float s = Sl[row * 32 + c];
                int col = mb2 + c;
                if (col == r0 || col == r1 || col == r2) s = -1.0e9f;
                sv[c] = s;
                cmax = fmaxf(cmax, s);
            }
            float mNew = fmaxf(mOld, cmax);
            float scale = __expf(mOld - mNew);
            float sum = 0.f;
#pragma unroll
            for (int c = 0; c < 32; ++c) {
                float p = __expf(sv[c] - mNew);
                sum += p;
                Pl[row * 32 + c] = f2bf(p);
            }
            rowSum[row] = rowSum[row] * scale + sum;
            rowM[row] = mNew;
            rowScale[row] = scale;
        }
        __syncthreads();

        {   // O = O*scale + P @ V  (this wave: 16 q rows x 32 d cols)
            FragBF aP;
            const __bf16* pp = &Pl[(qsub * 16 + lm) * 32 + (lh ? 0 : 8)];
            aP.h[0] = ldv8(pp);
            aP.h[1] = ldv8(pp + 16);
#pragma unroll
            for (int r = 0; r < 8; ++r) {
                int lr = qsub * 16 + r + (lh ? 0 : 8);
                float sc = rowScale[lr];
                oa0[r] *= sc;
                oa1[r] *= sc;
            }
            const int off0 = lh ? 0 : 8;
            FragBF bV0, bV1;
            const __bf16* v0p = &Vl[(dhalf * 32 + lm) * 32 + off0];
            bV0.h[0] = ldv8(v0p); bV0.h[1] = ldv8(v0p + 16);
            const __bf16* v1p = &Vl[(dhalf * 32 + 16 + lm) * 32 + off0];
            bV1.h[0] = ldv8(v1p); bV1.h[1] = ldv8(v1p + 16);
            oa0 = wmma_bf16(aP, bV0, oa0);
            oa1 = wmma_bf16(aP, bV1, oa1);
        }
    }
    __syncthreads();
#pragma unroll
    for (int r = 0; r < 8; ++r) {
        int lr = qsub * 16 + r + (lh ? 0 : 8);
        float inv = 1.f / rowSum[lr];
        int d0 = dhalf * 32 + lm;
        ao[(size_t)(qb + lr) * E_DIM + hh * 64 + d0]      = oa0[r] * inv;
        ao[(size_t)(qb + lr) * E_DIM + hh * 64 + d0 + 16] = oa1[r] * inv;
    }
}

// ---------------------------------------------------------------------------
// Dueling head: out[n][j] = sv + adv[j] - mean(adv).  One wave32 per row.
// ---------------------------------------------------------------------------
__global__ __launch_bounds__(32) void final_k(const float* __restrict__ attno,
                                              const float* __restrict__ advw,
                                              const float* __restrict__ advb,
                                              const float* __restrict__ stw,
                                              const float* __restrict__ stb,
                                              float* __restrict__ out) {
    const int n = blockIdx.x;
    const int lane = threadIdx.x;
    const float* x = attno + (size_t)n * E_DIM + lane * 8;
    float xs[8];
#pragma unroll
    for (int t = 0; t < 8; ++t) xs[t] = x[t];
    float s[6];
#pragma unroll
    for (int j = 0; j < 5; ++j) {
        float a = 0.f;
#pragma unroll
        for (int t = 0; t < 8; ++t) a += xs[t] * advw[j * E_DIM + lane * 8 + t];
        s[j] = a;
    }
    {
        float a = 0.f;
#pragma unroll
        for (int t = 0; t < 8; ++t) a += xs[t] * stw[lane * 8 + t];
        s[5] = a;
    }
#pragma unroll
    for (int j = 0; j < 6; ++j)
        for (int off = 16; off > 0; off >>= 1) s[j] += __shfl_xor(s[j], off);
    if (lane == 0) {
        float a[5], mean = 0.f;
#pragma unroll
        for (int j = 0; j < 5; ++j) { a[j] = s[j] + advb[j]; mean += a[j]; }
        mean *= 0.2f;
        float svv = s[5] + stb[0];
#pragma unroll
        for (int j = 0; j < 5; ++j) out[n * 5 + j] = svv + a[j] - mean;
    }
}

// ---------------------------------------------------------------------------
// Host orchestration
// ---------------------------------------------------------------------------
extern "C" void kernel_launch(void* const* d_in, const int* in_sizes, int n_in,
                              void* d_out, int out_size, void* d_ws, size_t ws_size,
                              hipStream_t stream) {
    (void)in_sizes; (void)n_in; (void)out_size; (void)ws_size;
    const float* obs      = (const float*)d_in[0];
    const float* pos      = (const float*)d_in[1];
    const float* conv1_w  = (const float*)d_in[2];
    const float* conv1_b  = (const float*)d_in[3];
    const float* r1a_w    = (const float*)d_in[4];
    const float* r1a_b    = (const float*)d_in[5];
    const float* r1b_w    = (const float*)d_in[6];
    const float* r1b_b    = (const float*)d_in[7];
    const float* r2a_w    = (const float*)d_in[8];
    const float* r2a_b    = (const float*)d_in[9];
    const float* r2b_w    = (const float*)d_in[10];
    const float* r2b_b    = (const float*)d_in[11];
    const float* conv2_w  = (const float*)d_in[12];
    const float* conv2_b  = (const float*)d_in[13];
    const float* obsfc_w  = (const float*)d_in[14];
    const float* obsfc_b  = (const float*)d_in[15];
    const float* posfc_w  = (const float*)d_in[16];
    const float* posfc_b  = (const float*)d_in[17];
    const float* pres_w1  = (const float*)d_in[18];
    const float* pres_b1  = (const float*)d_in[19];
    const float* pres_w2  = (const float*)d_in[20];
    const float* pres_b2  = (const float*)d_in[21];
    const float* cres_w1  = (const float*)d_in[22];
    const float* cres_b1  = (const float*)d_in[23];
    const float* cres_w2  = (const float*)d_in[24];
    const float* cres_b2  = (const float*)d_in[25];
    const float* gru_wih  = (const float*)d_in[26];
    const float* gru_bih  = (const float*)d_in[28];
    const float* gru_bhh  = (const float*)d_in[29];
    const float* attin_w  = (const float*)d_in[30];
    const float* attin_b  = (const float*)d_in[31];
    const float* attout_w = (const float*)d_in[32];
    const float* attout_b = (const float*)d_in[33];
    const float* adv_w    = (const float*)d_in[34];
    const float* adv_b    = (const float*)d_in[35];
    const float* state_w  = (const float*)d_in[36];
    const float* state_b  = (const float*)d_in[37];

    size_t off = 0;
    auto alloc = [&](size_t bytes) -> void* {
        off = (off + 255) & ~(size_t)255;
        void* p = (char*)d_ws + off;
        off += bytes;
        return p;
    };
    __bf16* wpk_c1   = (__bf16*)alloc((size_t)128 * 64 * 2);
    __bf16* wpk_r1a  = (__bf16*)alloc((size_t)128 * 1152 * 2);
    __bf16* wpk_r1b  = (__bf16*)alloc((size_t)128 * 1152 * 2);
    __bf16* wpk_r2a  = (__bf16*)alloc((size_t)128 * 1152 * 2);
    __bf16* wpk_r2b  = (__bf16*)alloc((size_t)128 * 1152 * 2);
    __bf16* wpk_obs  = (__bf16*)alloc((size_t)224 * 672 * 2);
    __bf16* wpk_pos  = (__bf16*)alloc((size_t)32 * 32 * 2);
    __bf16* wpk_p1   = (__bf16*)alloc((size_t)32 * 32 * 2);
    __bf16* wpk_p2   = (__bf16*)alloc((size_t)32 * 32 * 2);
    __bf16* wpk_cr1  = (__bf16*)alloc((size_t)256 * 256 * 2);
    __bf16* wpk_cr2  = (__bf16*)alloc((size_t)256 * 256 * 2);
    __bf16* wpk_gru  = (__bf16*)alloc((size_t)768 * 256 * 2);
    __bf16* wpk_ain  = (__bf16*)alloc((size_t)768 * 256 * 2);
    __bf16* wpk_aout = (__bf16*)alloc((size_t)256 * 256 * 2);
    __bf16* bufA     = (__bf16*)alloc((size_t)NN * 128 * PIX * 2);
    __bf16* bufB     = (__bf16*)alloc((size_t)NN * 128 * PIX * 2);
    float*  x2       = (float*)alloc((size_t)NN * 648 * 4);
    float*  lat      = (float*)alloc((size_t)NN * 256 * 4);
    float*  P0       = (float*)alloc((size_t)NN * 32 * 4);
    float*  P1       = (float*)alloc((size_t)NN * 32 * 4);
    float*  t1       = (float*)alloc((size_t)NN * 256 * 4);
    float*  gx       = (float*)alloc((size_t)NN * 768 * 4);
    float*  hbuf     = (float*)alloc((size_t)NN * 256 * 4);
    float*  qkv      = (float*)alloc((size_t)NN * 768 * 4);
    __bf16* qkvbf    = (__bf16*)alloc((size_t)NN * 768 * 2);
    int*    mc       = (int*)alloc((size_t)NN * 4 * 4);
    float*  ao       = (float*)alloc((size_t)NN * 256 * 4);
    float*  attno    = (float*)alloc((size_t)NN * 256 * 4);

    auto packblk = [](int rows, int kpad) { return (rows * kpad + 255) / 256; };
    pack_w_k<<<packblk(128, 64), 256, 0, stream>>>(conv1_w, wpk_c1, 128, 54, 64);
    pack_w_k<<<packblk(128, 1152), 256, 0, stream>>>(r1a_w, wpk_r1a, 128, 1152, 1152);
    pack_w_k<<<packblk(128, 1152), 256, 0, stream>>>(r1b_w, wpk_r1b, 128, 1152, 1152);
    pack_w_k<<<packblk(128, 1152), 256, 0, stream>>>(r2a_w, wpk_r2a, 128, 1152, 1152);
    pack_w_k<<<packblk(128, 1152), 256, 0, stream>>>(r2b_w, wpk_r2b, 128, 1152, 1152);
    pack_w_k<<<packblk(224, 672), 256, 0, stream>>>(obsfc_w, wpk_obs, 224, 648, 672);
    pack_w_k<<<packblk(32, 32), 256, 0, stream>>>(posfc_w, wpk_pos, 32, 4, 32);
    pack_w_k<<<packblk(32, 32), 256, 0, stream>>>(pres_w1, wpk_p1, 32, 32, 32);
    pack_w_k<<<packblk(32, 32), 256, 0, stream>>>(pres_w2, wpk_p2, 32, 32, 32);
    pack_w_k<<<packblk(256, 256), 256, 0, stream>>>(cres_w1, wpk_cr1, 256, 256, 256);
    pack_w_k<<<packblk(256, 256), 256, 0, stream>>>(cres_w2, wpk_cr2, 256, 256, 256);
    pack_w_k<<<packblk(768, 256), 256, 0, stream>>>(gru_wih, wpk_gru, 768, 256, 256);
    pack_w_k<<<packblk(768, 256), 256, 0, stream>>>(attin_w, wpk_ain, 768, 256, 256);
    pack_w_k<<<packblk(256, 256), 256, 0, stream>>>(attout_w, wpk_aout, 256, 256, 256);

    // conv stack (bf16 activations)
    conv3x3_k<6, 64, true, false, true><<<NN, 256, 0, stream>>>(
        obs, wpk_c1, conv1_b, (const __bf16*)nullptr, bufA);
    conv3x3_k<128, 1152, false, false, true><<<NN, 256, 0, stream>>>(
        bufA, wpk_r1a, r1a_b, (const __bf16*)nullptr, bufB);
    conv3x3_k<128, 1152, false, true, true><<<NN, 256, 0, stream>>>(
        bufB, wpk_r1b, r1b_b, bufA, bufA);
    conv3x3_k<128, 1152, false, false, true><<<NN, 256, 0, stream>>>(
        bufA, wpk_r2a, r2a_b, (const __bf16*)nullptr, bufB);
    conv3x3_k<128, 1152, false, true, true><<<NN, 256, 0, stream>>>(
        bufB, wpk_r2b, r2b_b, bufA, bufA);
    conv2_k<<<NN, 256, 0, stream>>>(bufA, conv2_w, conv2_b, x2);

    // obs_fc -> lat[:, 0:224]
    linear_k<<<dim3(64, 2), 256, 0, stream>>>(x2, 648, 648, 672, wpk_obs, obsfc_b,
                                              224, nullptr, 0, lat, 256, 1);
    // pos path -> lat[:, 224:256]
    linear_k<<<dim3(64, 1), 256, 0, stream>>>(pos, 4, 4, 32, wpk_pos, posfc_b,
                                              32, nullptr, 0, P0, 32, 1);
    linear_k<<<dim3(64, 1), 256, 0, stream>>>(P0, 32, 32, 32, wpk_p1, pres_b1,
                                              32, nullptr, 0, P1, 32, 1);
    linear_k<<<dim3(64, 1), 256, 0, stream>>>(P1, 32, 32, 32, wpk_p2, pres_b2,
                                              32, P0, 32, lat + 224, 256, 1);
    // concat residual block
    linear_k<<<dim3(64, 2), 256, 0, stream>>>(lat, 256, 256, 256, wpk_cr1, cres_b1,
                                              256, nullptr, 0, t1, 256, 1);
    linear_k<<<dim3(64, 2), 256, 0, stream>>>(t1, 256, 256, 256, wpk_cr2, cres_b2,
                                              256, lat, 256, lat, 256, 1);
    // GRU (h0 = 0)
    linear_k<<<dim3(64, 6), 256, 0, stream>>>(lat, 256, 256, 256, wpk_gru, gru_bih,
                                              768, nullptr, 0, gx, 768, 0);
    gru_k<<<(NN * 256 + 255) / 256, 256, 0, stream>>>(gx, gru_bhh, hbuf, NN * 256);
    // attention
    linear_k<<<dim3(64, 6), 256, 0, stream>>>(hbuf, 256, 256, 256, wpk_ain, attin_b,
                                              768, nullptr, 0, qkv, 768, 0);
    cvt_bf_k<<<(NN * 768 + 255) / 256, 256, 0, stream>>>(qkv, qkvbf, NN * 768);
    mask_k<<<NN, 256, 0, stream>>>(pos, mc);
    attn_k<<<dim3(64, 4), 256, 0, stream>>>(qkvbf, mc, ao);
    linear_k<<<dim3(64, 2), 256, 0, stream>>>(ao, 256, 256, 256, wpk_aout, attout_b,
                                              256, nullptr, 0, attno, 256, 0);
    // dueling head
    final_k<<<NN, 32, 0, stream>>>(attno, adv_w, adv_b, state_w, state_b,
                                   (float*)d_out);
}
